// REFINEModule_40733469835941
// MI455X (gfx1250) — matrix operand order repaired and checked by reading, not compiled, not run on hardware
//
#include <hip/hip_runtime.h>
#include <math.h>

// ---------------- types ----------------
typedef __bf16 bf16;
typedef __bf16 v16bf __attribute__((ext_vector_type(16)));
typedef float  v8f   __attribute__((ext_vector_type(8)));

// ---------------- problem constants ----------------
// D=15, N = 2^16-1 = 65535 nodes. Internal nodes 0..32766, leaves 32767..65534.
#define NNODES 65535
#define NI     32767      // number of internal nodes == index of first leaf
#define NLEAF  32768

// ---------------- swizzled bf16 weight buffer offsets (elements) ----------------
#define W_OP1  0u
#define W_TB1  2048u
#define W_FT1  4096u
#define W_JN1  6144u
#define W_OP2  10240u
#define W_TB2  14336u
#define W_FT2  18432u
#define W_JN2  22528u
#define W_XOU  26624u
#define W_BTOA 223232u
#define W_CA   305152u
#define W_CB   370688u
#define W_AB   436224u
#define W_O1   501760u
#define W_END  534528u          // bf16 elements -> 1,069,056 bytes (256B aligned)

// f32 workspace region (float offsets from start of d_ws)
#define FBASE   267264u
#define ROWI    (32768u*256u)
#define XX_OFF  (FBASE)
#define FF_OFF  (XX_OFF + ROWI)
#define RR_OFF  (FF_OFF + ROWI)
#define XA_OFF  (RR_OFF + ROWI)
#define HH_OFF  (XA_OFF + ROWI)
#define CC_OFF  (HH_OFF + ROWI)  // 65536 x 256 floats

__device__ __forceinline__ float sigmoidf_(float x) { return 1.0f / (1.0f + expf(-x)); }

__device__ __forceinline__ v8f wmma_bf16(v16bf a, v16bf b, v8f c) {
    return __builtin_amdgcn_wmma_f32_16x16x32_bf16(false, a, false, b, (short)0, c, false, false);
}

// A fragment (16x32 bf16, MxK) from row-major bf16 in LDS.
// ISA layout: lane (r = lane&15, hi = lane>>4); element e -> K = (e>=8?16:0) + hi*8 + (e&7)
__device__ __forceinline__ v16bf load_a(const bf16* A, int stride, int lane, int kBase) {
    int r = lane & 15, hi = lane >> 4;
    const bf16* p = A + r * stride + kBase + hi * 8;
    v16bf a;
#pragma unroll
    for (int e = 0; e < 8; ++e) { a[e] = p[e]; a[e + 8] = p[16 + e]; }
    return a;
}

// B fragment (32x16 bf16, KxN) from the pre-swizzled global weight buffer:
// one aligned 32-byte load per lane.
__device__ __forceinline__ v16bf load_b(const bf16* __restrict__ W, int nTiles, int kt, int nt, int lane) {
    return *reinterpret_cast<const v16bf*>(W + ((size_t)((kt * nTiles + nt) * 32 + lane) << 4));
}

// ---------------- weight prep: f32 row-major -> fragment-swizzled bf16 (zero K-padded) --
__global__ void prep_weights(const float* __restrict__ src, bf16* __restrict__ dst,
                             int K, int N, int Kpad) {
    int idx = blockIdx.x * blockDim.x + threadIdx.x;
    int total = Kpad * N;
    if (idx >= total) return;
    int e = idx & 15, lane = (idx >> 4) & 31, frag = idx >> 9;
    int nTiles = N >> 4;
    int kt = frag / nTiles, nt = frag - kt * nTiles;
    int k = kt * 32 + ((e >> 3) << 4) + (lane >> 4) * 8 + (e & 7);
    int n = nt * 16 + (lane & 15);
    float v = (k < K) ? src[(size_t)k * N + n] : 0.0f;
    dst[idx] = (bf16)v;
}

// ---------------- internal nodes (M=64 per wave): 4 MLPs -> x -> xou -------------------
__global__ __launch_bounds__(32) void refine_internal(
    const float* __restrict__ oper, const float* __restrict__ tbf,
    const float* __restrict__ ftf,  const float* __restrict__ jnf,
    const float* __restrict__ b_op1, const float* __restrict__ b_op2,
    const float* __restrict__ b_tb1, const float* __restrict__ b_tb2,
    const float* __restrict__ b_ft1, const float* __restrict__ b_ft2,
    const float* __restrict__ b_jn1, const float* __restrict__ b_jn2,
    const float* __restrict__ b_xou, const bf16* __restrict__ wbuf,
    float* __restrict__ XX, float* __restrict__ FF,
    float* __restrict__ RR, float* __restrict__ XA)
{
    __shared__ bf16 actA[64 * 64];
    __shared__ bf16 mid [64 * 64];
    __shared__ bf16 xbf [64 * 256];
    const int lane = threadIdx.x;
    const int base = blockIdx.x * 64;
    const int c = lane & 15, hi = lane >> 4;

#pragma unroll
    for (int f = 0; f < 4; ++f) {
        const float* feat = (f == 0) ? oper : (f == 1) ? tbf : (f == 2) ? ftf : jnf;
        const float* b1   = (f == 0) ? b_op1 : (f == 1) ? b_tb1 : (f == 2) ? b_ft1 : b_jn1;
        const float* b2   = (f == 0) ? b_op2 : (f == 1) ? b_tb2 : (f == 2) ? b_ft2 : b_jn2;
        const bf16*  w1   = wbuf + ((f == 0) ? W_OP1 : (f == 1) ? W_TB1 : (f == 2) ? W_FT1 : W_JN1);
        const bf16*  w2   = wbuf + ((f == 0) ? W_OP2 : (f == 1) ? W_TB2 : (f == 2) ? W_FT2 : W_JN2);
        const int K  = (f == 0) ? 4 : (f == 1) ? 11 : (f == 2) ? 18 : 37;
        const int KP = (f == 3) ? 64 : 32;

        for (int idx = lane; idx < 64 * KP; idx += 32) {
            int r = idx / KP, k = idx - r * KP;
            float v = (k < K) ? feat[(size_t)(base + r) * K + k] : 0.0f;
            actA[r * KP + k] = (bf16)v;
        }
        __syncthreads();
        // layer 1: [64,KP] @ [KP,64]
        for (int nt = 0; nt < 4; ++nt) {
            v8f acc[4] = {};
            for (int kt = 0; kt < (KP >> 5); ++kt) {
                v16bf b = load_b(w1, 4, kt, nt, lane);
#pragma unroll
                for (int rt = 0; rt < 4; ++rt)
                    acc[rt] = wmma_bf16(load_a(actA + rt * 16 * KP, KP, lane, kt * 32), b, acc[rt]);
            }
            float bv = b1[nt * 16 + c];
#pragma unroll
            for (int rt = 0; rt < 4; ++rt)
#pragma unroll
                for (int j = 0; j < 8; ++j) {
                    int m = rt * 16 + j + 8 * hi;
                    mid[m * 64 + nt * 16 + c] = (bf16)fmaxf(acc[rt][j] + bv, 0.0f);
                }
        }
        __syncthreads();
        // layer 2: [64,64] @ [64,64] -> slice of x
        for (int nt = 0; nt < 4; ++nt) {
            v8f acc[4] = {};
            for (int kt = 0; kt < 2; ++kt) {
                v16bf b = load_b(w2, 4, kt, nt, lane);
#pragma unroll
                for (int rt = 0; rt < 4; ++rt)
                    acc[rt] = wmma_bf16(load_a(mid + rt * 16 * 64, 64, lane, kt * 32), b, acc[rt]);
            }
            float bv = b2[nt * 16 + c];
            int col = f * 64 + nt * 16 + c;
#pragma unroll
            for (int rt = 0; rt < 4; ++rt)
#pragma unroll
                for (int j = 0; j < 8; ++j) {
                    int m = rt * 16 + j + 8 * hi;
                    float v = fmaxf(acc[rt][j] + bv, 0.0f);
                    xbf[m * 256 + col] = (bf16)v;
                    int node = base + m;
                    if (node < NI) XA[(size_t)node * 256 + col] = v;
                }
        }
        __syncthreads();
    }
    // xou: [64,256] @ [256,768]  (48 N-tiles x 8 K-steps, 4 row-tiles per B fragment)
    const bf16* wx = wbuf + W_XOU;
    for (int nt = 0; nt < 48; ++nt) {
        if (nt < 47) __builtin_prefetch(wx + ((size_t)((nt + 1) * 32) << 4), 0, 0);
        v8f acc[4] = {};
        for (int kt = 0; kt < 8; ++kt) {
            v16bf b = load_b(wx, 48, kt, nt, lane);
#pragma unroll
            for (int rt = 0; rt < 4; ++rt)
                acc[rt] = wmma_bf16(load_a(xbf + rt * 16 * 256, 256, lane, kt * 32), b, acc[rt]);
        }
        float bv = b_xou[nt * 16 + c];
        int seg = nt >> 4, col = (nt & 15) * 16 + c;
#pragma unroll
        for (int rt = 0; rt < 4; ++rt)
#pragma unroll
            for (int j = 0; j < 8; ++j) {
                int m = rt * 16 + j + 8 * hi, node = base + m;
                if (node >= NI) continue;
                float v = acc[rt][j] + bv;
                size_t o = (size_t)node * 256 + col;
                if (seg == 0)      XX[o] = v;
                else if (seg == 1) FF[o] = sigmoidf_(v);
                else               RR[o] = sigmoidf_(v);
            }
    }
}

// ---------------- leaves (M=32 per wave): cb; fused wa/wb -> u; c_ab -> C[leaf] --------
__global__ __launch_bounds__(32) void refine_leaf(
    const float* __restrict__ ca,     const float* __restrict__ cbm,
    const float* __restrict__ b_btoa, const float* __restrict__ b_ca,
    const float* __restrict__ b_cb,   const float* __restrict__ b_ab,
    const bf16* __restrict__ wbuf,    float* __restrict__ CC)
{
    __shared__ bf16 cabf[32 * 256];   // bf16 c_a
    __shared__ bf16 cbin[32 * 320];   // bf16 c_b; reused as u buffer
    __shared__ bf16 cbbf[32 * 256];   // bf16 cb
    const int lane = threadIdx.x;
    const int base = blockIdx.x * 32;
    const int c = lane & 15, hi = lane >> 4;

    for (int idx = lane; idx < 32 * 256; idx += 32) {
        int r = idx >> 8, k = idx & 255;
        cabf[idx] = (bf16)ca[(size_t)(NI + base + r) * 256 + k];
    }
    for (int idx = lane; idx < 32 * 320; idx += 32) {
        int r = idx / 320, k = idx - r * 320;
        cbin[idx] = (bf16)cbm[(size_t)(NI + base + r) * 320 + k];
    }
    __syncthreads();
    // cb = c_b @ w_btoa + b   (K=320 -> 10 K-steps)
    for (int nt = 0; nt < 16; ++nt) {
        v8f acc[2] = {};
        for (int kt = 0; kt < 10; ++kt) {
            v16bf b = load_b(wbuf + W_BTOA, 16, kt, nt, lane);
#pragma unroll
            for (int rt = 0; rt < 2; ++rt)
                acc[rt] = wmma_bf16(load_a(cbin + rt * 16 * 320, 320, lane, kt * 32), b, acc[rt]);
        }
        float bv = b_btoa[nt * 16 + c];
#pragma unroll
        for (int rt = 0; rt < 2; ++rt)
#pragma unroll
            for (int j = 0; j < 8; ++j) {
                int m = rt * 16 + j + 8 * hi;
                cbbf[m * 256 + nt * 16 + c] = (bf16)(acc[rt][j] + bv);
            }
    }
    __syncthreads();
    // fused gates: wa = sig(c_a@w_ca+b), wb = sig(cb@w_cb+b), u = wa*c_a + wb*cb
    bf16* ubf = cbin;   // c_b staging no longer needed
    for (int nt = 0; nt < 16; ++nt) {
        v8f accA[2] = {}, accB[2] = {};
        for (int kt = 0; kt < 8; ++kt) {
            v16bf bA = load_b(wbuf + W_CA, 16, kt, nt, lane);
            v16bf bB = load_b(wbuf + W_CB, 16, kt, nt, lane);
#pragma unroll
            for (int rt = 0; rt < 2; ++rt) {
                accA[rt] = wmma_bf16(load_a(cabf + rt * 16 * 256, 256, lane, kt * 32), bA, accA[rt]);
                accB[rt] = wmma_bf16(load_a(cbbf + rt * 16 * 256, 256, lane, kt * 32), bB, accB[rt]);
            }
        }
        float bva = b_ca[nt * 16 + c], bvb = b_cb[nt * 16 + c];
#pragma unroll
        for (int rt = 0; rt < 2; ++rt)
#pragma unroll
            for (int j = 0; j < 8; ++j) {
                int m = rt * 16 + j + 8 * hi, col = nt * 16 + c;
                float wa = sigmoidf_(accA[rt][j] + bva);
                float wb = sigmoidf_(accB[rt][j] + bvb);
                float u = wa * (float)cabf[m * 256 + col] + wb * (float)cbbf[m * 256 + col];
                ubf[m * 256 + col] = (bf16)u;
            }
    }
    __syncthreads();
    // c_ab = relu(u @ w_ab + b) -> C[leaf]
    for (int nt = 0; nt < 16; ++nt) {
        v8f acc[2] = {};
        for (int kt = 0; kt < 8; ++kt) {
            v16bf b = load_b(wbuf + W_AB, 16, kt, nt, lane);
#pragma unroll
            for (int rt = 0; rt < 2; ++rt)
                acc[rt] = wmma_bf16(load_a(ubf + rt * 16 * 256, 256, lane, kt * 32), b, acc[rt]);
        }
        float bv = b_ab[nt * 16 + c];
#pragma unroll
        for (int rt = 0; rt < 2; ++rt)
#pragma unroll
            for (int j = 0; j < 8; ++j) {
                int m = rt * 16 + j + 8 * hi, node = NI + base + m;
                CC[(size_t)node * 256 + nt * 16 + c] = fmaxf(acc[rt][j] + bv, 0.0f);
            }
    }
}

// ---------------- one tree level: c[p] = ff*(c[2p+1]+c[2p+2]) + (1-ff)*xx --------------
__global__ void level_kernel(float* __restrict__ CC, float* __restrict__ HH,
                             const float* __restrict__ XX, const float* __restrict__ FF,
                             const float* __restrict__ RR, const float* __restrict__ XA,
                             int first, int count) {
    int i = blockIdx.x * blockDim.x + threadIdx.x;
    if (i >= count * 256) return;
    int p = first + (i >> 8);
    int d = i & 255;
    size_t pi = (size_t)p * 256 + d;
    float csum = CC[(size_t)(2 * p + 1) * 256 + d] + CC[(size_t)(2 * p + 2) * 256 + d];
    float ffv = FF[pi], xxv = XX[pi], rrv = RR[pi], xv = XA[pi];
    float cv = ffv * csum + (1.0f - ffv) * xxv;
    CC[pi] = cv;
    HH[pi] = rrv * tanhf(cv) + (1.0f - rrv) * xv;
}

// ---------------- output head (M=32): sigmoid(relu(h@w_o1+b)@w_o2+b) ------------------
__global__ __launch_bounds__(32) void refine_out(
    const float* __restrict__ HH, const bf16* __restrict__ wbuf,
    const float* __restrict__ b_o1, const float* __restrict__ w_o2,
    const float* __restrict__ b_o2, float* __restrict__ out)
{
    __shared__ bf16  hbf [32 * 256];
    __shared__ float hidf[32 * 128];
    const int lane = threadIdx.x;
    const int base = blockIdx.x * 32;
    const int c = lane & 15, hi = lane >> 4;

    for (int idx = lane; idx < 32 * 256; idx += 32) {
        int r = idx >> 8, k = idx & 255;
        int node = base + r;
        float v = (node < NI) ? HH[(size_t)node * 256 + k] : 0.0f;
        hbf[idx] = (bf16)v;
    }
    __syncthreads();
    for (int nt = 0; nt < 8; ++nt) {
        v8f acc[2] = {};
        for (int kt = 0; kt < 8; ++kt) {
            v16bf b = load_b(wbuf + W_O1, 8, kt, nt, lane);
#pragma unroll
            for (int rt = 0; rt < 2; ++rt)
                acc[rt] = wmma_bf16(load_a(hbf + rt * 16 * 256, 256, lane, kt * 32), b, acc[rt]);
        }
        float bv = b_o1[nt * 16 + c];
#pragma unroll
        for (int rt = 0; rt < 2; ++rt)
#pragma unroll
            for (int j = 0; j < 8; ++j) {
                int m = rt * 16 + j + 8 * hi;
                hidf[m * 128 + nt * 16 + c] = fmaxf(acc[rt][j] + bv, 0.0f);
            }
    }
    __syncthreads();
    // final matvec: one 128-dot per lane (32 rows == 32 lanes)
    float s = 0.0f;
#pragma unroll 8
    for (int j = 0; j < 128; ++j) s += hidf[lane * 128 + j] * w_o2[j];
    int node = base + lane;
    if (node < NI) out[node] = sigmoidf_(s + b_o2[0]);
}

// ---------------- leaves share one output scalar: sigmoid(relu(b_o1)·w_o2 + b_o2) -----
__global__ void leaf_out_kernel(const float* __restrict__ b_o1, const float* __restrict__ w_o2,
                                const float* __restrict__ b_o2, float* __restrict__ out) {
    __shared__ float red[128];
    int t = threadIdx.x; // 128 threads
    red[t] = fmaxf(b_o1[t], 0.0f) * w_o2[t];
    __syncthreads();
    for (int s = 64; s > 0; s >>= 1) {
        if (t < s) red[t] += red[t + s];
        __syncthreads();
    }
    float val = sigmoidf_(red[0] + b_o2[0]);
    for (int i = t; i < NLEAF; i += 128) out[NI + i] = val;
}

// ---------------- host launch ----------------
extern "C" void kernel_launch(void* const* d_in, const int* in_sizes, int n_in,
                              void* d_out, int out_size, void* d_ws, size_t ws_size,
                              hipStream_t stream) {
    (void)in_sizes; (void)n_in; (void)out_size; (void)ws_size;
    bf16*  wbuf = (bf16*)d_ws;
    float* fws  = (float*)d_ws;
    float* XX = fws + XX_OFF; float* FF = fws + FF_OFF; float* RR = fws + RR_OFF;
    float* XA = fws + XA_OFF; float* HH = fws + HH_OFF; float* CC = fws + CC_OFF;
    float* out = (float*)d_out;

    // 1) weight swizzle f32 -> fragment-layout bf16
    struct PW { const void* src; unsigned off; int K, N, Kpad; };
    const PW pw[14] = {
        {d_in[6],  W_OP1,  4,   64,  32}, {d_in[10], W_TB1, 11,  64,  32},
        {d_in[14], W_FT1, 18,   64,  32}, {d_in[18], W_JN1, 37,  64,  64},
        {d_in[8],  W_OP2, 64,   64,  64}, {d_in[12], W_TB2, 64,  64,  64},
        {d_in[16], W_FT2, 64,   64,  64}, {d_in[20], W_JN2, 64,  64,  64},
        {d_in[22], W_XOU, 256, 768, 256}, {d_in[26], W_BTOA,320, 256, 320},
        {d_in[28], W_CA,  256, 256, 256}, {d_in[30], W_CB,  256, 256, 256},
        {d_in[24], W_AB,  256, 256, 256}, {d_in[32], W_O1,  256, 128, 256},
    };
    for (int i = 0; i < 14; ++i) {
        int total = pw[i].Kpad * pw[i].N;
        prep_weights<<<(total + 255) / 256, 256, 0, stream>>>(
            (const float*)pw[i].src, wbuf + pw[i].off, pw[i].K, pw[i].N, pw[i].Kpad);
    }

    // 2) internal-node GEMMs (512 waves x 64 nodes)
    refine_internal<<<512, 32, 0, stream>>>(
        (const float*)d_in[0], (const float*)d_in[1], (const float*)d_in[2], (const float*)d_in[3],
        (const float*)d_in[7], (const float*)d_in[9], (const float*)d_in[11], (const float*)d_in[13],
        (const float*)d_in[15], (const float*)d_in[17], (const float*)d_in[19], (const float*)d_in[21],
        (const float*)d_in[23], wbuf, XX, FF, RR, XA);

    // 3) leaf-cell GEMMs (1024 waves x 32 leaves)
    refine_leaf<<<1024, 32, 0, stream>>>(
        (const float*)d_in[4], (const float*)d_in[5],
        (const float*)d_in[27], (const float*)d_in[29], (const float*)d_in[31], (const float*)d_in[25],
        wbuf, CC);

    // 4) tree recursion, level by level (complete tree: children of p are 2p+1, 2p+2)
    for (int n = 1; n <= 15; ++n) {
        int count = 1 << (15 - n);
        int first = count - 1;
        int threads = count * 256;
        level_kernel<<<(threads + 255) / 256, 256, 0, stream>>>(CC, HH, XX, FF, RR, XA, first, count);
    }

    // 5) output head for internal nodes + shared leaf scalar
    refine_out<<<1024, 32, 0, stream>>>(HH, wbuf,
        (const float*)d_in[33], (const float*)d_in[34], (const float*)d_in[35], out);
    leaf_out_kernel<<<1, 128, 0, stream>>>(
        (const float*)d_in[33], (const float*)d_in[34], (const float*)d_in[35], out);
}
